// GATConv_dgl_67508295958892
// MI455X (gfx1250) — matrix operand order repaired and checked by reading, not compiled, not run on hardware
//
#include <hip/hip_runtime.h>

#define N_NODES   50000
#define N_EDGES   800000
#define IN_FEATS  256
#define OUT_FEATS 64
#define NUM_HEADS 4
#define HF        (OUT_FEATS * NUM_HEADS)   // 256
#define NEG_SLOPE 0.2f

typedef __attribute__((ext_vector_type(16))) __bf16 v16bf;
typedef __attribute__((ext_vector_type(8)))  __bf16 v8bf;
typedef __attribute__((ext_vector_type(8)))  float  v8f;

// ---------------------------------------------------------------------------
// f32 -> bf16 conversion (feat and W)
// ---------------------------------------------------------------------------
__global__ void cvt_bf16_kernel(const float* __restrict__ in,
                                __bf16* __restrict__ out, int n) {
    int i = blockIdx.x * blockDim.x + threadIdx.x;
    if (i < n) out[i] = (__bf16)in[i];
}

// ---------------------------------------------------------------------------
// H = feat_bf @ W_bf^T  via v_wmma_f32_16x16x32_bf16.
// One wave computes a 16-row x 64-col strip (4 WMMA accumulators).
// A layout (16-bit A 16x32): lane<16 -> M=lane, K pairs {0..7,16..23};
//                            lane>=16 -> M=lane-16, K pairs {8..15,24..31}.
// B layout (32x16): lane<16 -> N=lane, K {0..7,16..23}; lane>=16 -> N=lane-16, K {8..15,24..31}.
// Both map to contiguous bf16 runs in row-major feat / W.
// ---------------------------------------------------------------------------
__global__ __launch_bounds__(256) void gemm_wmma_kernel(
    const __bf16* __restrict__ A,   // [N_NODES, IN_FEATS] row-major
    const __bf16* __restrict__ Wb,  // [HF, IN_FEATS] row-major (W as given)
    float* __restrict__ H)          // [N_NODES, HF]
{
    const int M_TILES = N_NODES / 16;          // 3125 (exact)
    const int NG      = HF / 64;               // 4 groups of 64 cols
    const int wave    = blockIdx.x * (blockDim.x >> 5) + (threadIdx.x >> 5);
    if (wave >= M_TILES * NG) return;

    const int mt   = wave / NG;
    const int ng   = wave % NG;
    const int row0 = mt * 16;
    const int col0 = ng * 64;

    const int lane    = threadIdx.x & 31;
    const int l15     = lane & 15;
    const int halfsel = lane >> 4;

    const int am  = row0 + l15;        // A row for this lane
    const int ako = halfsel * 8;       // A K sub-offset
    const int bko = halfsel * 8;       // B K sub-offset

    v8f acc[4] = {};

    for (int kb = 0; kb < IN_FEATS; kb += 32) {
        // Prefetch next A chunk (speculative, dropped if OOB)
        __builtin_prefetch(A + (size_t)am * IN_FEATS + kb + 64, 0, 0);

        union { v16bf v; v8bf h[2]; } au;
        au.h[0] = *(const v8bf*)(A + (size_t)am * IN_FEATS + kb + ako);
        au.h[1] = *(const v8bf*)(A + (size_t)am * IN_FEATS + kb + ako + 16);
        const v16bf a = au.v;

        #pragma unroll
        for (int j = 0; j < 4; ++j) {
            const int bn = col0 + 16 * j + l15;   // output column (row of W)
            union { v16bf v; v8bf h[2]; } bu;
            bu.h[0] = *(const v8bf*)(Wb + (size_t)bn * IN_FEATS + kb + bko);
            bu.h[1] = *(const v8bf*)(Wb + (size_t)bn * IN_FEATS + kb + bko + 16);
            const v16bf b = bu.v;
            acc[j] = __builtin_amdgcn_wmma_f32_16x16x32_bf16(
                false, a, false, b, (short)0, acc[j], false, false);
        }
    }

    // C/D layout: VGPR v -> M = v + 8*halfsel, N = l15
    const int mo = halfsel * 8;
    #pragma unroll
    for (int j = 0; j < 4; ++j) {
        const int n = col0 + 16 * j + l15;
        #pragma unroll
        for (int v = 0; v < 8; ++v) {
            H[(size_t)(row0 + mo + v) * HF + n] = acc[j][v];
        }
    }
}

// ---------------------------------------------------------------------------
// el/er per (node, head): dot(h[n,hd,:], attn_{l,r}[hd,:])
// ---------------------------------------------------------------------------
__global__ void node_logits_kernel(const float* __restrict__ H,
                                   const float* __restrict__ attn_l,
                                   const float* __restrict__ attn_r,
                                   float* __restrict__ el,
                                   float* __restrict__ er) {
    int idx = blockIdx.x * blockDim.x + threadIdx.x;
    if (idx >= N_NODES * NUM_HEADS) return;
    const int n  = idx / NUM_HEADS;
    const int hd = idx % NUM_HEADS;
    const float* hp = H + (size_t)n * HF + hd * OUT_FEATS;
    const float* al = attn_l + hd * OUT_FEATS;
    const float* ar = attn_r + hd * OUT_FEATS;
    float sl = 0.f, sr = 0.f;
    #pragma unroll 8
    for (int f = 0; f < OUT_FEATS; ++f) {
        const float v = hp[f];
        sl += v * al[f];
        sr += v * ar[f];
    }
    el[idx] = sl;
    er[idx] = sr;
}

// ---------------------------------------------------------------------------
// Initialize emax = -inf, esum = 0, out = 0 (harness poisons buffers)
// ---------------------------------------------------------------------------
__global__ void init_kernel(float* __restrict__ emax, float* __restrict__ esum,
                            float* __restrict__ out) {
    int i = blockIdx.x * blockDim.x + threadIdx.x;
    if (i < N_NODES * NUM_HEADS) {
        emax[i] = -__builtin_inff();
        esum[i] = 0.f;
    }
    if (i < N_NODES * HF) out[i] = 0.f;
}

// ---------------------------------------------------------------------------
// Pass 1: segment max over dst (native f32 atomic max -> global_atomic_max_num_f32)
// ---------------------------------------------------------------------------
__global__ void edge_max_kernel(const int* __restrict__ src, const int* __restrict__ dst,
                                const float* __restrict__ el, const float* __restrict__ er,
                                float* __restrict__ emax) {
    int i = blockIdx.x * blockDim.x + threadIdx.x;
    if (i >= N_EDGES * NUM_HEADS) return;
    const int e  = i / NUM_HEADS;
    const int hd = i % NUM_HEADS;
    const int s = src[e], d = dst[e];
    float v = el[s * NUM_HEADS + hd] + er[d * NUM_HEADS + hd];
    v = v > 0.f ? v : NEG_SLOPE * v;
    atomicMax(&emax[d * NUM_HEADS + hd], v);
}

// ---------------------------------------------------------------------------
// Pass 2: ex = exp(e - emax[dst]); segment-sum into esum (f32 atomic add)
// ---------------------------------------------------------------------------
__global__ void edge_exp_kernel(const int* __restrict__ src, const int* __restrict__ dst,
                                const float* __restrict__ el, const float* __restrict__ er,
                                const float* __restrict__ emax,
                                float* __restrict__ esum, float* __restrict__ ex) {
    int i = blockIdx.x * blockDim.x + threadIdx.x;
    if (i >= N_EDGES * NUM_HEADS) return;
    const int e  = i / NUM_HEADS;
    const int hd = i % NUM_HEADS;
    const int s = src[e], d = dst[e];
    float v = el[s * NUM_HEADS + hd] + er[d * NUM_HEADS + hd];
    v = v > 0.f ? v : NEG_SLOPE * v;
    const float x = __expf(v - emax[d * NUM_HEADS + hd]);
    ex[i] = x;
    atomicAdd(&esum[d * NUM_HEADS + hd], x);
}

// ---------------------------------------------------------------------------
// Pass 3: out[dst] += h[src] * alpha.  One wave32 per edge; each lane moves
// 8 contiguous floats (float4 x2 gather, 8 native f32 atomic adds).
// ---------------------------------------------------------------------------
__global__ __launch_bounds__(256) void edge_scatter_kernel(
    const int* __restrict__ src, const int* __restrict__ dst,
    const float* __restrict__ H, const float* __restrict__ ex,
    const float* __restrict__ esum, float* __restrict__ out) {
    const int wave = blockIdx.x * (blockDim.x >> 5) + (threadIdx.x >> 5);
    if (wave >= N_EDGES) return;
    const int lane = threadIdx.x & 31;
    const int s = src[wave], d = dst[wave];
    const int c0 = lane * 8;            // 8 feats per lane; 8 | 64 so one head
    const int hd = c0 >> 6;
    const float alpha = ex[wave * NUM_HEADS + hd] / esum[d * NUM_HEADS + hd];
    const float* hsrc = H + (size_t)s * HF + c0;
    float* od = out + (size_t)d * HF + c0;
    const float4 v0 = *(const float4*)(hsrc);
    const float4 v1 = *(const float4*)(hsrc + 4);
    atomicAdd(&od[0], v0.x * alpha);
    atomicAdd(&od[1], v0.y * alpha);
    atomicAdd(&od[2], v0.z * alpha);
    atomicAdd(&od[3], v0.w * alpha);
    atomicAdd(&od[4], v1.x * alpha);
    atomicAdd(&od[5], v1.y * alpha);
    atomicAdd(&od[6], v1.z * alpha);
    atomicAdd(&od[7], v1.w * alpha);
}

// ---------------------------------------------------------------------------
extern "C" void kernel_launch(void* const* d_in, const int* in_sizes, int n_in,
                              void* d_out, int out_size, void* d_ws, size_t ws_size,
                              hipStream_t stream) {
    const float* feat   = (const float*)d_in[0];
    const float* W      = (const float*)d_in[1];
    const float* attn_l = (const float*)d_in[2];
    const float* attn_r = (const float*)d_in[3];
    const int*   src    = (const int*)d_in[4];
    const int*   dst    = (const int*)d_in[5];
    float* out = (float*)d_out;

    // Workspace carve-out (~93 MB total)
    char* ws = (char*)d_ws;
    size_t off = 0;
    auto carve = [&](size_t bytes) -> void* {
        void* p = ws + off;
        off += (bytes + 255) & ~(size_t)255;
        return p;
    };
    __bf16* feat_bf = (__bf16*)carve((size_t)N_NODES * IN_FEATS * sizeof(__bf16));
    __bf16* W_bf    = (__bf16*)carve((size_t)HF * IN_FEATS * sizeof(__bf16));
    float*  H       = (float*)carve((size_t)N_NODES * HF * sizeof(float));
    float*  el      = (float*)carve((size_t)N_NODES * NUM_HEADS * sizeof(float));
    float*  er      = (float*)carve((size_t)N_NODES * NUM_HEADS * sizeof(float));
    float*  emax    = (float*)carve((size_t)N_NODES * NUM_HEADS * sizeof(float));
    float*  esum    = (float*)carve((size_t)N_NODES * NUM_HEADS * sizeof(float));
    float*  ex      = (float*)carve((size_t)N_EDGES * NUM_HEADS * sizeof(float));

    const int BLK = 256;

    // 1) convert feat and W to bf16
    {
        const int n = N_NODES * IN_FEATS;
        cvt_bf16_kernel<<<(n + BLK - 1) / BLK, BLK, 0, stream>>>(feat, feat_bf, n);
    }
    {
        const int n = HF * IN_FEATS;
        cvt_bf16_kernel<<<(n + BLK - 1) / BLK, BLK, 0, stream>>>(W, W_bf, n);
    }

    // 2) WMMA GEMM: H = feat @ W^T
    {
        const int waves  = (N_NODES / 16) * (HF / 64);   // 12500
        const int blocks = (waves + 7) / 8;              // 8 waves per block
        gemm_wmma_kernel<<<blocks, BLK, 0, stream>>>(feat_bf, W_bf, H);
    }

    // 3) per-node logits
    {
        const int n = N_NODES * NUM_HEADS;
        node_logits_kernel<<<(n + BLK - 1) / BLK, BLK, 0, stream>>>(H, attn_l, attn_r, el, er);
    }

    // 4) init emax/esum/out
    {
        const int n = N_NODES * HF;
        init_kernel<<<(n + BLK - 1) / BLK, BLK, 0, stream>>>(emax, esum, out);
    }

    // 5) edge softmax pass 1: segment max
    {
        const int n = N_EDGES * NUM_HEADS;
        edge_max_kernel<<<(n + BLK - 1) / BLK, BLK, 0, stream>>>(src, dst, el, er, emax);
    }

    // 6) edge softmax pass 2: exp + segment sum
    {
        const int n = N_EDGES * NUM_HEADS;
        edge_exp_kernel<<<(n + BLK - 1) / BLK, BLK, 0, stream>>>(src, dst, el, er, emax, esum, ex);
    }

    // 7) message passing scatter: one wave per edge
    {
        const int blocks = (N_EDGES + 7) / 8;
        edge_scatter_kernel<<<blocks, BLK, 0, stream>>>(src, dst, H, ex, esum, out);
    }
}
